// SimplifiedAttention_2954937500240
// MI455X (gfx1250) — matrix operand order repaired and checked by reading, not compiled, not run on hardware
//
#include <hip/hip_runtime.h>
#include <hip/hip_bf16.h>

typedef _Float16 half_t;
typedef __attribute__((ext_vector_type(16))) _Float16 v16h;
typedef __attribute__((ext_vector_type(8)))  _Float16 v8h;
typedef __attribute__((ext_vector_type(8)))  float    v8f;

#define D_MODEL 1024
#define N_HEADS 16
#define D_K     64
#define BATCH   4
#define SEQ     2048
#define MROWS   (BATCH*SEQ)   // 8192

union AB { v16h v; v8h h[2]; };

static __device__ __forceinline__ v8f wmma_f16(const AB& a, const AB& b, v8f c) {
    return __builtin_amdgcn_wmma_f32_16x16x32_f16(false, a.v, false, b.v,
                                                  (short)0, c, false, false);
}

// ---------------------------------------------------------------------------
// f32 -> f16 conversion (grid-stride)
// ---------------------------------------------------------------------------
__global__ void cvt_kernel(const float* __restrict__ src, half_t* __restrict__ dst, int n) {
    int i = blockIdx.x * blockDim.x + threadIdx.x;
    int stride = gridDim.x * blockDim.x;
    for (; i < n; i += stride) dst[i] = (half_t)src[i];
}

// ---------------------------------------------------------------------------
// Projection GEMM: Out = X(f16)[M,K] @ W(f16)[N,K]^T + bias, f16 output.
// One wave per 32(M) x 64(N) tile: each B fragment feeds 2 WMMAs.
// vmode 0: write [B,H,L,64]   vmode 1: write [B,H,64,L] (transposed V)
// ---------------------------------------------------------------------------
__global__ void __launch_bounds__(32)
proj_gemm_kernel(const half_t* __restrict__ X,
                 const half_t* __restrict__ W,
                 const float*  __restrict__ bias,
                 half_t* __restrict__ out, int vmode) {
    const int lane = threadIdx.x & 31;
    const int l15  = lane & 15;
    const int hsel = (lane >> 4) & 1;
    const int m0 = blockIdx.x * 32;
    const int n0 = blockIdx.y * 64;

    v8f acc[2][4] = {};
    const half_t* pA0 = X + (long)(m0 + l15) * D_MODEL;
    const half_t* pA1 = X + (long)(m0 + 16 + l15) * D_MODEL;

    #pragma unroll 2
    for (int k0 = 0; k0 < D_MODEL; k0 += 32) {
        AB a0, a1;
        a0.h[0] = *(const v8h*)(pA0 + k0 + hsel * 8);
        a0.h[1] = *(const v8h*)(pA0 + k0 + hsel * 8 + 16);
        a1.h[0] = *(const v8h*)(pA1 + k0 + hsel * 8);
        a1.h[1] = *(const v8h*)(pA1 + k0 + hsel * 8 + 16);
        #pragma unroll
        for (int j = 0; j < 4; ++j) {
            const half_t* pB = W + (long)(n0 + j * 16 + l15) * D_MODEL + k0 + hsel * 8;
            AB b;
            b.h[0] = *(const v8h*)(pB);
            b.h[1] = *(const v8h*)(pB + 16);
            acc[0][j] = wmma_f16(a0, b, acc[0][j]);
            acc[1][j] = wmma_f16(a1, b, acc[1][j]);
        }
    }

    #pragma unroll
    for (int i = 0; i < 2; ++i) {
        #pragma unroll
        for (int j = 0; j < 4; ++j) {
            const int col = n0 + j * 16 + l15;
            const float bv = bias[col];
            const int hh = col >> 6, d = col & 63;
            #pragma unroll
            for (int r = 0; r < 8; ++r) {
                const int row = m0 + i * 16 + r + hsel * 8;
                const int bb = row >> 11, l = row & (SEQ - 1);
                const float v = acc[i][j][r] + bv;
                long idx;
                if (vmode == 0)
                    idx = (((long)(bb * N_HEADS + hh) * SEQ + l) * D_K) + d;
                else
                    idx = (((long)(bb * N_HEADS + hh) * D_K + d) * SEQ) + l;
                out[idx] = (half_t)v;
            }
        }
    }
}

// ---------------------------------------------------------------------------
// Output GEMM: out(f32) = Ctx(f16)[M,K] @ Wo(f16)[N,K]^T + bo   (32x64 tile)
// ---------------------------------------------------------------------------
__global__ void __launch_bounds__(32)
out_gemm_kernel(const half_t* __restrict__ X,
                const half_t* __restrict__ W,
                const float*  __restrict__ bias,
                float* __restrict__ out) {
    const int lane = threadIdx.x & 31;
    const int l15  = lane & 15;
    const int hsel = (lane >> 4) & 1;
    const int m0 = blockIdx.x * 32;
    const int n0 = blockIdx.y * 64;

    v8f acc[2][4] = {};
    const half_t* pA0 = X + (long)(m0 + l15) * D_MODEL;
    const half_t* pA1 = X + (long)(m0 + 16 + l15) * D_MODEL;

    #pragma unroll 2
    for (int k0 = 0; k0 < D_MODEL; k0 += 32) {
        AB a0, a1;
        a0.h[0] = *(const v8h*)(pA0 + k0 + hsel * 8);
        a0.h[1] = *(const v8h*)(pA0 + k0 + hsel * 8 + 16);
        a1.h[0] = *(const v8h*)(pA1 + k0 + hsel * 8);
        a1.h[1] = *(const v8h*)(pA1 + k0 + hsel * 8 + 16);
        #pragma unroll
        for (int j = 0; j < 4; ++j) {
            const half_t* pB = W + (long)(n0 + j * 16 + l15) * D_MODEL + k0 + hsel * 8;
            AB b;
            b.h[0] = *(const v8h*)(pB);
            b.h[1] = *(const v8h*)(pB + 16);
            acc[0][j] = wmma_f16(a0, b, acc[0][j]);
            acc[1][j] = wmma_f16(a1, b, acc[1][j]);
        }
    }

    #pragma unroll
    for (int i = 0; i < 2; ++i) {
        #pragma unroll
        for (int j = 0; j < 4; ++j) {
            const int col = n0 + j * 16 + l15;
            const float bv = bias[col];
            #pragma unroll
            for (int r = 0; r < 8; ++r) {
                const int row = m0 + i * 16 + r + hsel * 8;
                out[(long)row * D_MODEL + col] = acc[i][j][r] + bv;
            }
        }
    }
}

// ---------------------------------------------------------------------------
// Flash-style attention.  One wave per 32 query rows of one (b,h); K/V block
// fragments loaded once per 32-key step and reused by both 16-row q tiles.
//   Qh, Kh : [B,H,L,64] f16     Vth : [B,H,64,L] f16 (transposed)
//   Ctx    : [B,L,1024] f16
// ---------------------------------------------------------------------------
__global__ void __launch_bounds__(32)
attn_kernel(const half_t* __restrict__ Qh,
            const half_t* __restrict__ Kh,
            const half_t* __restrict__ Vth,
            half_t* __restrict__ Ctx) {
    __shared__ __align__(16) _Float16 plds[2][16][40];   // per-qtile 16x32 P

    const int lane = threadIdx.x & 31;
    const int l15  = lane & 15;
    const int hsel = (lane >> 4) & 1;
    const int bh = blockIdx.y;
    const int b  = bh >> 4, h = bh & 15;
    const int q0 = blockIdx.x * 32;

    const half_t* Qb = Qh  + (long)bh * SEQ * D_K;
    const half_t* Kb = Kh  + (long)bh * SEQ * D_K;
    const half_t* Vb = Vth + (long)bh * D_K * SEQ;

    // Q fragments: 2 q-tiles x 2 k-steps, resident for whole pass
    AB qa[2][2];
    #pragma unroll
    for (int qt = 0; qt < 2; ++qt) {
        const half_t* pQ = Qb + (long)(q0 + qt * 16 + l15) * D_K;
        #pragma unroll
        for (int t = 0; t < 2; ++t) {
            qa[qt][t].h[0] = *(const v8h*)(pQ + t * 32 + hsel * 8);
            qa[qt][t].h[1] = *(const v8h*)(pQ + t * 32 + hsel * 8 + 16);
        }
    }

    v8f ctx[2][4] = {};
    float mr[2][8], lr[2][8];
    #pragma unroll
    for (int qt = 0; qt < 2; ++qt)
        #pragma unroll
        for (int r = 0; r < 8; ++r) { mr[qt][r] = -1.0e30f; lr[qt][r] = 0.0f; }

    for (int l0 = 0; l0 < SEQ; l0 += 32) {
        // ---- K block fragments (shared by both q tiles) ----
        AB kb[2][2];   // [k-step t][key half]
        #pragma unroll
        for (int t = 0; t < 2; ++t) {
            const int d0 = t * 32 + hsel * 8;
            const half_t* p0 = Kb + (long)(l0 + l15) * D_K + d0;
            const half_t* p1 = Kb + (long)(l0 + 16 + l15) * D_K + d0;
            kb[t][0].h[0] = *(const v8h*)(p0); kb[t][0].h[1] = *(const v8h*)(p0 + 16);
            kb[t][1].h[0] = *(const v8h*)(p1); kb[t][1].h[1] = *(const v8h*)(p1 + 16);
        }
        // ---- V block fragments (shared by both q tiles) ----
        AB vbf[4];
        #pragma unroll
        for (int j = 0; j < 4; ++j) {
            const half_t* pV = Vb + (long)(j * 16 + l15) * SEQ + l0 + hsel * 8;
            vbf[j].h[0] = *(const v8h*)(pV);
            vbf[j].h[1] = *(const v8h*)(pV + 16);
        }

        #pragma unroll
        for (int qt = 0; qt < 2; ++qt) {
            // ---- scores: S[16 x 32] = Q_qt @ K_block^T ----
            v8f s0 = {}, s1 = {};
            #pragma unroll
            for (int t = 0; t < 2; ++t) {
                s0 = wmma_f16(qa[qt][t], kb[t][0], s0);
                s1 = wmma_f16(qa[qt][t], kb[t][1], s1);
            }

            // ---- online softmax (rows live in 16-lane groups = C halves) ----
            #pragma unroll
            for (int r = 0; r < 8; ++r) {
                float a0 = fminf(fmaxf(s0[r] * 0.125f, -100.0f), 100.0f);
                float a1 = fminf(fmaxf(s1[r] * 0.125f, -100.0f), 100.0f);
                float mx = fmaxf(a0, a1);
                #pragma unroll
                for (int off = 1; off < 16; off <<= 1)
                    mx = fmaxf(mx, __shfl_xor(mx, off, 32));
                const float mnew  = fmaxf(mr[qt][r], mx);
                const float alpha = __expf(mr[qt][r] - mnew);
                const float p0v = __expf(a0 - mnew);
                const float p1v = __expf(a1 - mnew);
                float rs = p0v + p1v;
                #pragma unroll
                for (int off = 1; off < 16; off <<= 1)
                    rs += __shfl_xor(rs, off, 32);
                lr[qt][r] = lr[qt][r] * alpha + rs;
                mr[qt][r] = mnew;
                #pragma unroll
                for (int j = 0; j < 4; ++j) ctx[qt][j][r] *= alpha;
                const int row = r + hsel * 8;
                plds[qt][row][l15]      = (half_t)p0v;   // C-layout -> LDS
                plds[qt][row][16 + l15] = (half_t)p1v;
            }
            asm volatile("s_wait_dscnt 0" ::: "memory");

            // ---- reload P in A-layout, then ctx += P @ V ----
            AB pa;
            pa.h[0] = *(const v8h*)(&plds[qt][l15][hsel * 8]);
            pa.h[1] = *(const v8h*)(&plds[qt][l15][hsel * 8 + 16]);
            #pragma unroll
            for (int j = 0; j < 4; ++j)
                ctx[qt][j] = wmma_f16(pa, vbf[j], ctx[qt][j]);
        }
    }

    // ---- normalize + store ctx as [B, L, 1024] f16 ----
    #pragma unroll
    for (int qt = 0; qt < 2; ++qt) {
        #pragma unroll
        for (int j = 0; j < 4; ++j) {
            const int col = h * 64 + j * 16 + l15;
            #pragma unroll
            for (int r = 0; r < 8; ++r) {
                const int row = q0 + qt * 16 + r + hsel * 8;
                const float v = ctx[qt][j][r] / lr[qt][r];
                Ctx[((long)(b * SEQ + row)) * D_MODEL + col] = (half_t)v;
            }
        }
    }
}

// ---------------------------------------------------------------------------
extern "C" void kernel_launch(void* const* d_in, const int* in_sizes, int n_in,
                              void* d_out, int out_size, void* d_ws, size_t ws_size,
                              hipStream_t stream) {
    (void)in_sizes; (void)n_in; (void)out_size; (void)ws_size;
    const float* x  = (const float*)d_in[0];
    const float* Wq = (const float*)d_in[1];
    const float* bq = (const float*)d_in[2];
    const float* Wk = (const float*)d_in[3];
    const float* bk = (const float*)d_in[4];
    const float* Wv = (const float*)d_in[5];
    const float* bv = (const float*)d_in[6];
    const float* Wo = (const float*)d_in[7];
    const float* bo = (const float*)d_in[8];
    float* out = (float*)d_out;

    char* ws = (char*)d_ws;
    const size_t MB = 1024 * 1024;
    half_t* x_h  = (half_t*)(ws + 0);        // 16 MB
    half_t* Wq_h = (half_t*)(ws + 16 * MB);  //  2 MB
    half_t* Wk_h = (half_t*)(ws + 18 * MB);
    half_t* Wv_h = (half_t*)(ws + 20 * MB);
    half_t* Wo_h = (half_t*)(ws + 22 * MB);
    half_t* Q_h  = (half_t*)(ws + 24 * MB);  // 16 MB  [B,H,L,64]
    half_t* K_h  = (half_t*)(ws + 40 * MB);  // 16 MB  [B,H,L,64]
    half_t* Vt_h = (half_t*)(ws + 56 * MB);  // 16 MB  [B,H,64,L]
    half_t* C_h  = (half_t*)(ws + 72 * MB);  // 16 MB  [B,L,1024]

    const int NX = MROWS * D_MODEL;
    const int NW = D_MODEL * D_MODEL;
    cvt_kernel<<<2048, 256, 0, stream>>>(x,  x_h,  NX);
    cvt_kernel<<<512,  256, 0, stream>>>(Wq, Wq_h, NW);
    cvt_kernel<<<512,  256, 0, stream>>>(Wk, Wk_h, NW);
    cvt_kernel<<<512,  256, 0, stream>>>(Wv, Wv_h, NW);
    cvt_kernel<<<512,  256, 0, stream>>>(Wo, Wo_h, NW);

    dim3 g(MROWS / 32, D_MODEL / 64);   // 256 x 16 tiles of 32x64
    proj_gemm_kernel<<<g, 32, 0, stream>>>(x_h, Wq_h, bq, Q_h, 0);
    proj_gemm_kernel<<<g, 32, 0, stream>>>(x_h, Wk_h, bk, K_h, 0);
    proj_gemm_kernel<<<g, 32, 0, stream>>>(x_h, Wv_h, bv, Vt_h, 1);

    attn_kernel<<<dim3(SEQ / 32, BATCH * N_HEADS), 32, 0, stream>>>(Q_h, K_h, Vt_h, C_h);

    out_gemm_kernel<<<g, 32, 0, stream>>>(C_h, Wo_h, bo, out);
}
